// ParticleNet_16114717294919
// MI455X (gfx1250) — matrix operand order repaired and checked by reading, not compiled, not run on hardware
//
#include <hip/hip_runtime.h>
#include <hip/hip_bf16.h>

#define BATCH 256
#define NPTS  128
#define NFEAT 16
#define KNBR  16
#define EPS_BN 1e-3f

typedef __attribute__((ext_vector_type(16))) _Float16 v16h;
typedef __attribute__((ext_vector_type(8)))  float    v8f;

union FragU { v16h v; unsigned int u[8]; };

// ---------------------------------------------------------------------------
// WMMA fragment loaders.
// B-matrix (32x16 K x N, f16): prepacked in fragment-major order so each lane
// reads its 16 halves (32B) contiguously -> 2x global_load_b128.
// Lane L holds column N = L&15, K {0..7,16..23} (L<16) or {8..15,24..31} (L>=16),
// half j -> K = base + j + (j<8 ? 0 : 8), base = 0 or 8.  (Packing kernel below matches this exactly.)
// ---------------------------------------------------------------------------
__device__ __forceinline__ v16h load_bfrag(const _Float16* __restrict__ p, int lane) {
  FragU f;
  const uint4* q = (const uint4*)(p + lane * 16);
  uint4 a = q[0];
  uint4 b = q[1];
  f.u[0] = a.x; f.u[1] = a.y; f.u[2] = a.z; f.u[3] = a.w;
  f.u[4] = b.x; f.u[5] = b.y; f.u[6] = b.z; f.u[7] = b.w;
  return f.v;
}

// A-matrix (16x32 M x K, f16) from row-major LDS tile.
// ISA layout: lanes 0-15 row M=lane, K in {0..7,16..23}; lanes 16-31 same rows,
// K in {8..15,24..31}; K pairs contiguous -> 8x ds_load_b32 per fragment.
__device__ __forceinline__ v16h load_afrag(const _Float16* x, int lane, int kbase, int stride) {
  FragU f;
  const int M    = lane & 15;
  const int kofs = (lane < 16) ? 0 : 8;
  const _Float16* row = x + M * stride + kbase + kofs;
#pragma unroll
  for (int vv = 0; vv < 8; ++vv) {
    const int K0 = (vv < 4) ? (2 * vv) : (2 * vv + 8);
    f.u[vv] = *(const unsigned int*)(row + K0);
  }
  return f.v;
}

// D = X(16 x KTOT) * W(KTOT x NT*16), f32 accumulate.
template <int KTOT, int NT>
__device__ __forceinline__ void wmma_mm(const _Float16* X, const _Float16* __restrict__ wp,
                                        v8f (&acc)[NT], int lane) {
  constexpr int KT = KTOT / 32;
#pragma unroll 1
  for (int kt = 0; kt < KT; ++kt) {
    if (kt + 1 < KT)  // pull next weight K-slab toward L2/L0 (global_prefetch_b8)
      __builtin_prefetch((const void*)(wp + (size_t)((kt + 1) * NT) * 512 + lane * 16), 0, 0);
    v16h a = load_afrag(X, lane, kt * 32, KTOT);
#pragma unroll
    for (int nt = 0; nt < NT; ++nt) {
      v16h b = load_bfrag(wp + (size_t)(kt * NT + nt) * 512, lane);
      acc[nt] = __builtin_amdgcn_wmma_f32_16x16x32_f16(
          /*neg_a=*/false, a, /*neg_b=*/false, b,
          /*c_mod=*/(short)0, acc[nt], /*reuse_a=*/false, /*reuse_b=*/false);
    }
  }
}

template <int NT>
__device__ __forceinline__ void zero_acc(v8f (&acc)[NT]) {
#pragma unroll
  for (int n = 0; n < NT; ++n)
#pragma unroll
    for (int j = 0; j < 8; ++j) acc[n][j] = 0.f;
}

// Folded-BN + ReLU, write f16 activation tile back to LDS (next layer's A).
// D layout: vgpr r -> M = r (lanes 0-15) or 8+r (lanes 16-31); N = lane&15.
template <int NT>
__device__ __forceinline__ void bn_relu_store(const v8f (&acc)[NT], const float* __restrict__ sb,
                                              _Float16* Y, int lane) {
  constexpr int COUTL = NT * 16;
  const int N = lane & 15;
  const int mofs = (lane < 16) ? 0 : 8;
#pragma unroll
  for (int nt = 0; nt < NT; ++nt) {
    const int ch = nt * 16 + N;
    const float s = sb[ch];
    const float t = sb[COUTL + ch];
#pragma unroll
    for (int r = 0; r < 8; ++r) {
      float v = fmaf(acc[nt][r], s, t);
      Y[(mofs + r) * COUTL + ch] = (_Float16)((v > 0.f) ? v : 0.f);
    }
  }
}

// ---------------------------------------------------------------------------
// Prep: points (pt*cos f1, pt*sin f1), mask, coord shift, BN0(features) -> fts0 (B,P,16)
// ---------------------------------------------------------------------------
__global__ __launch_bounds__(256) void prep_kernel(
    const float* __restrict__ feat, const float* __restrict__ g, const float* __restrict__ be,
    const float* __restrict__ mu, const float* __restrict__ var,
    float* __restrict__ points, float* __restrict__ shift, float* __restrict__ mask,
    float* __restrict__ fts0) {
  const int i = blockIdx.x * blockDim.x + threadIdx.x;
  if (i >= BATCH * NPTS) return;
  const float* __restrict__ f = feat + (size_t)i * NFEAT;
  const float pt = f[0];
  points[i * 2 + 0] = pt * cosf(f[1]);
  points[i * 2 + 1] = pt * sinf(f[1]);
  float red = 0.f;
#pragma unroll
  for (int c = 0; c < NFEAT; ++c) red += f[c];
  const float m = (red != 0.f) ? 1.f : 0.f;
  mask[i] = m;
  shift[i] = 1.0e9f * (1.f - m);
  float* __restrict__ o = fts0 + (size_t)i * NFEAT;
#pragma unroll
  for (int c = 0; c < NFEAT; ++c)
    o[c] = (f[c] - mu[c]) * g[c] * rsqrtf(var[c] + EPS_BN) + be[c];
}

// ---------------------------------------------------------------------------
// Fold BN into per-channel scale/bias:  y = x*s + t
// ---------------------------------------------------------------------------
__global__ void bnfold_kernel(const float* __restrict__ g, const float* __restrict__ b,
                              const float* __restrict__ m, const float* __restrict__ v,
                              float* __restrict__ sb, int C) {
  const int c = blockIdx.x * blockDim.x + threadIdx.x;
  if (c >= C) return;
  const float s = g[c] * rsqrtf(v[c] + EPS_BN);
  sb[c] = s;
  sb[C + c] = b[c] - m[c] * s;
}

// ---------------------------------------------------------------------------
// Pack W (Cin x Cout, f32, row-major) into f16 WMMA B-fragments, K padded
// to a multiple of 32 with zeros (needed for the 16-ch block-0 shortcut).
// ---------------------------------------------------------------------------
__global__ void pack_w_kernel(const float* __restrict__ W, _Float16* __restrict__ dst,
                              int Cin, int Cout) {
  const int i = blockIdx.x * blockDim.x + threadIdx.x;
  const int NT = Cout / 16;
  const int Kpad = (Cin + 31) & ~31;
  const int total = Kpad * Cout;
  if (i >= total) return;
  const int j = i & 15;
  const int L = (i >> 4) & 31;
  const int frag = i >> 9;
  const int nt = frag % NT;
  const int kt = frag / NT;
  const int cin = kt * 32 + ((L < 16) ? 0 : 8) + ((j < 8) ? j : (j + 8));
  const int cout = nt * 16 + (L & 15);
  dst[i] = (cin < Cin) ? (_Float16)W[(size_t)cin * Cout + cout] : (_Float16)0.f;
}

// ---------------------------------------------------------------------------
// KNN: one block per batch, one thread per point. Coords staged in LDS
// (shifted by per-point coord_shift), top-(K+1) insertion select, drop self.
// ---------------------------------------------------------------------------
template <int C>
__global__ __launch_bounds__(128) void knn_kernel(const float* __restrict__ pts,
                                                  const float* __restrict__ shift,
                                                  int* __restrict__ knn) {
  __shared__ float Ps[NPTS * C];
  __shared__ float rr[NPTS];
  const int b = blockIdx.x;
  const int p = threadIdx.x;
  const float sh = shift[b * NPTS + p];
  const float* __restrict__ src = pts + ((size_t)b * NPTS + p) * C;
  float my[C];
  float r = 0.f;
#pragma unroll
  for (int c = 0; c < C; ++c) {
    const float v = src[c] + sh;
    my[c] = v;
    Ps[p * C + c] = v;
    r = fmaf(v, v, r);
  }
  rr[p] = r;
  __syncthreads();
  float bd[KNBR + 1];
  int bi[KNBR + 1];
#pragma unroll
  for (int k = 0; k <= KNBR; ++k) { bd[k] = 3.0e38f; bi[k] = 0; }
  for (int q = 0; q < NPTS; ++q) {
    float dot = 0.f;
#pragma unroll 4
    for (int c = 0; c < C; ++c) dot = fmaf(my[c], Ps[q * C + c], dot);
    const float d = r + rr[q] - 2.f * dot;
    if (d < bd[KNBR]) {           // strict '<' keeps earlier index on ties (jax top_k)
      int k = KNBR;
      while (k > 0 && d < bd[k - 1]) { bd[k] = bd[k - 1]; bi[k] = bi[k - 1]; --k; }
      bd[k] = d; bi[k] = q;
    }
  }
  int* __restrict__ dst = knn + ((size_t)b * NPTS + p) * KNBR;
#pragma unroll
  for (int k = 0; k < KNBR; ++k) dst[k] = bi[k + 1];  // drop self
}

// ---------------------------------------------------------------------------
// Shortcut: sc = BN(fts @ Wsc). One wave per 16-point tile via WMMA.
// ---------------------------------------------------------------------------
template <int CINL, int COUT>
__global__ __launch_bounds__(128) void shortcut_kernel(const float* __restrict__ fts,
                                                       const _Float16* __restrict__ wsc,
                                                       const float* __restrict__ sb,
                                                       float* __restrict__ scb) {
  constexpr int CPAD = (CINL < 32) ? 32 : CINL;
  constexpr int NT = COUT / 16;
  __shared__ __align__(16) _Float16 xs[4][16 * 128];
  const int wave = threadIdx.x >> 5;
  const int lane = threadIdx.x & 31;
  const int p0 = (blockIdx.x * 4 + wave) * 16;
  _Float16* X = &xs[wave][0];
  const int row = lane & 15;
  const int hgrp = lane >> 4;
  const float* __restrict__ src = fts + (size_t)(p0 + row) * CINL;
  for (int c = hgrp * (CPAD / 2); c < (hgrp + 1) * (CPAD / 2); ++c)
    X[row * CPAD + c] = (c < CINL) ? (_Float16)src[c] : (_Float16)0.f;
  __syncthreads();
  v8f acc[NT];
  zero_acc<NT>(acc);
  wmma_mm<CPAD, NT>(X, wsc, acc, lane);
  const int N = lane & 15;
  const int mofs = (lane < 16) ? 0 : 8;
#pragma unroll
  for (int nt = 0; nt < NT; ++nt) {
    const int ch = nt * 16 + N;
    const float s = sb[ch];
    const float t = sb[COUT + ch];
#pragma unroll
    for (int r = 0; r < 8; ++r)
      scb[(size_t)(p0 + mofs + r) * COUT + ch] = fmaf(acc[nt][r], s, t);
  }
}

// ---------------------------------------------------------------------------
// EdgeConv: one wave per point. Gather K=16 edge rows [center | knn-center]
// into LDS (f16), chain 3 WMMA convs (BN folded + ReLU), mean over K via
// lane-half shfl, add shortcut, ReLU, write fts_out.
// ---------------------------------------------------------------------------
template <int CIN, int COUT>
__global__ __launch_bounds__(128) void edgeconv_kernel(
    const float* __restrict__ fts, const int* __restrict__ knn, const float* __restrict__ scb,
    const _Float16* __restrict__ w0, const float* __restrict__ sb0,
    const _Float16* __restrict__ w1, const float* __restrict__ sb1,
    const _Float16* __restrict__ w2, const float* __restrict__ sb2,
    float* __restrict__ out) {
  constexpr int CIN2 = 2 * CIN;
  constexpr int NT = COUT / 16;
  __shared__ __align__(16) _Float16 xs[4][2][16 * 256];  // ping/pong per wave
  const int wave = threadIdx.x >> 5;
  const int lane = threadIdx.x & 31;
  const int pt = blockIdx.x * 4 + wave;  // flat (b*P + p)
  const int b = pt >> 7;                 // P = 128
  _Float16* X = &xs[wave][0][0];
  _Float16* Y = &xs[wave][1][0];

  // gather: row = neighbor k; lane pair (row, row+16) splits the channel range
  const int row = lane & 15;
  const int hgrp = lane >> 4;
  const float* __restrict__ cen = fts + (size_t)pt * CIN;
  const int nb = knn[(size_t)pt * KNBR + row];
  const float* __restrict__ nf = fts + ((size_t)(b * NPTS + nb)) * CIN;
  const int c0 = hgrp * (CIN / 2);
#pragma unroll
  for (int c4 = 0; c4 < CIN / 8; ++c4) {
    const int c = c0 + c4 * 4;
    const float4 cv = *(const float4*)(cen + c);
    const float4 nv = *(const float4*)(nf + c);
    X[row * CIN2 + c + 0] = (_Float16)cv.x;
    X[row * CIN2 + c + 1] = (_Float16)cv.y;
    X[row * CIN2 + c + 2] = (_Float16)cv.z;
    X[row * CIN2 + c + 3] = (_Float16)cv.w;
    X[row * CIN2 + CIN + c + 0] = (_Float16)(nv.x - cv.x);
    X[row * CIN2 + CIN + c + 1] = (_Float16)(nv.y - cv.y);
    X[row * CIN2 + CIN + c + 2] = (_Float16)(nv.z - cv.z);
    X[row * CIN2 + CIN + c + 3] = (_Float16)(nv.w - cv.w);
  }
  __syncthreads();

  {  // conv0: (16 x CIN2) @ (CIN2 x COUT)
    v8f acc[NT];
    zero_acc<NT>(acc);
    wmma_mm<CIN2, NT>(X, w0, acc, lane);
    bn_relu_store<NT>(acc, sb0, Y, lane);
  }
  __syncthreads();
  {  // conv1
    v8f acc[NT];
    zero_acc<NT>(acc);
    wmma_mm<COUT, NT>(Y, w1, acc, lane);
    bn_relu_store<NT>(acc, sb1, X, lane);
  }
  __syncthreads();
  {  // conv2 + mean over K + shortcut + ReLU
    v8f acc[NT];
    zero_acc<NT>(acc);
    wmma_mm<COUT, NT>(X, w2, acc, lane);
    const int N = lane & 15;
    const float* __restrict__ scrow = scb + (size_t)pt * COUT;
    float* __restrict__ orow = out + (size_t)pt * COUT;
#pragma unroll
    for (int nt = 0; nt < NT; ++nt) {
      const int ch = nt * 16 + N;
      const float s = sb2[ch];
      const float t = sb2[COUT + ch];
      float sum = 0.f;
#pragma unroll
      for (int r = 0; r < 8; ++r) {  // ReLU before the K-mean (matches ref)
        const float v = fmaf(acc[nt][r], s, t);
        sum += (v > 0.f) ? v : 0.f;
      }
      sum += __shfl_xor(sum, 16, 32);  // combine rows 0..7 with 8..15 (wave32)
      if (lane < 16) {
        const float v = scrow[ch] + sum * (1.f / 16.f);
        orow[ch] = (v > 0.f) ? v : 0.f;
      }
    }
  }
}

// ---------------------------------------------------------------------------
// Masked mean over points, then dense+ReLU, 6-way logits, softmax.
// ---------------------------------------------------------------------------
__global__ __launch_bounds__(256) void pool_kernel(const float* __restrict__ fts,
                                                   const float* __restrict__ mask,
                                                   float* __restrict__ pool) {
  const int b = blockIdx.x;
  const int c = threadIdx.x;
  float s = 0.f;
  for (int p = 0; p < NPTS; ++p)
    s += fts[((size_t)b * NPTS + p) * 256 + c] * mask[b * NPTS + p];
  pool[b * 256 + c] = s * (1.f / (float)NPTS);
}

__global__ __launch_bounds__(256) void head_kernel(const float* __restrict__ pool,
                                                   const float* __restrict__ Wd,
                                                   const float* __restrict__ bd,
                                                   const float* __restrict__ Wo,
                                                   const float* __restrict__ bo,
                                                   float* __restrict__ out) {
  __shared__ float pl[256];
  __shared__ float xr[256];
  __shared__ float logits[6];
  const int b = blockIdx.x;
  const int t = threadIdx.x;
  pl[t] = pool[b * 256 + t];
  __syncthreads();
  float a = bd[t];
  for (int i = 0; i < 256; ++i) a = fmaf(pl[i], Wd[i * 256 + t], a);
  xr[t] = (a > 0.f) ? a : 0.f;
  __syncthreads();
  if (t < 6) {
    float a2 = bo[t];
    for (int i = 0; i < 256; ++i) a2 = fmaf(xr[i], Wo[i * 6 + t], a2);
    logits[t] = a2;
  }
  __syncthreads();
  if (t == 0) {
    float mx = -3.4e38f;
    for (int k = 0; k < 6; ++k) mx = fmaxf(mx, logits[k]);
    float e[6], s = 0.f;
    for (int k = 0; k < 6; ++k) { e[k] = expf(logits[k] - mx); s += e[k]; }
    const float inv = 1.f / s;
    for (int k = 0; k < 6; ++k) out[b * 6 + k] = e[k] * inv;
  }
}

// ---------------------------------------------------------------------------
static void* ws_take(char* base, size_t& off, size_t bytes) {
  void* p = base + off;
  off += (bytes + 255) & ~(size_t)255;
  return p;
}

extern "C" void kernel_launch(void* const* d_in, const int* in_sizes, int n_in,
                              void* d_out, int out_size, void* d_ws, size_t ws_size,
                              hipStream_t stream) {
  (void)in_sizes; (void)n_in; (void)out_size; (void)ws_size;
  const float* features = (const float*)d_in[0];
  char* ws = (char*)d_ws;
  size_t off = 0;
  const int NP = BATCH * NPTS;

  float* points = (float*)ws_take(ws, off, (size_t)NP * 2 * 4);
  float* shift  = (float*)ws_take(ws, off, (size_t)NP * 4);
  float* mask   = (float*)ws_take(ws, off, (size_t)NP * 4);
  float* ftsA   = (float*)ws_take(ws, off, (size_t)NP * 256 * 4);
  float* ftsB   = (float*)ws_take(ws, off, (size_t)NP * 256 * 4);
  float* scb    = (float*)ws_take(ws, off, (size_t)NP * 256 * 4);
  int*   knnb   = (int*)  ws_take(ws, off, (size_t)NP * KNBR * 4);
  float* pool   = (float*)ws_take(ws, off, (size_t)BATCH * 256 * 4);

  // conv metadata: {weight input idx, bn idx start, Cin, Cout}
  struct CM { int wi, bn, cin, cout; };
  const CM cm[12] = {
      {5, 6, 32, 64},    {10, 11, 64, 64},  {15, 16, 64, 64},  {20, 21, 16, 64},    // blk0 c0,c1,c2,sc
      {25, 26, 128, 128},{30, 31, 128, 128},{35, 36, 128, 128},{40, 41, 64, 128},   // blk1
      {45, 46, 256, 256},{50, 51, 256, 256},{55, 56, 256, 256},{60, 61, 128, 256}}; // blk2
  _Float16* pw[12];
  float* sb[12];
  for (int i = 0; i < 12; ++i) {
    const int kpad = (cm[i].cin + 31) & ~31;
    const size_t tot = (size_t)kpad * cm[i].cout;
    pw[i] = (_Float16*)ws_take(ws, off, tot * 2);
    sb[i] = (float*)ws_take(ws, off, (size_t)2 * cm[i].cout * 4);
    pack_w_kernel<<<(int)((tot + 255) / 256), 256, 0, stream>>>(
        (const float*)d_in[cm[i].wi], pw[i], cm[i].cin, cm[i].cout);
    bnfold_kernel<<<(cm[i].cout + 255) / 256, 256, 0, stream>>>(
        (const float*)d_in[cm[i].bn], (const float*)d_in[cm[i].bn + 1],
        (const float*)d_in[cm[i].bn + 2], (const float*)d_in[cm[i].bn + 3],
        sb[i], cm[i].cout);
  }

  prep_kernel<<<(NP + 255) / 256, 256, 0, stream>>>(
      features, (const float*)d_in[1], (const float*)d_in[2],
      (const float*)d_in[3], (const float*)d_in[4], points, shift, mask, ftsA);

  // block 0: coords = points (pt*cos, pt*sin)
  knn_kernel<2><<<BATCH, NPTS, 0, stream>>>(points, shift, knnb);
  shortcut_kernel<16, 64><<<NP / 64, 128, 0, stream>>>(ftsA, pw[3], sb[3], scb);
  edgeconv_kernel<16, 64><<<NP / 4, 128, 0, stream>>>(
      ftsA, knnb, scb, pw[0], sb[0], pw[1], sb[1], pw[2], sb[2], ftsB);

  // block 1: coords = fts (64 ch)
  knn_kernel<64><<<BATCH, NPTS, 0, stream>>>(ftsB, shift, knnb);
  shortcut_kernel<64, 128><<<NP / 64, 128, 0, stream>>>(ftsB, pw[7], sb[7], scb);
  edgeconv_kernel<64, 128><<<NP / 4, 128, 0, stream>>>(
      ftsB, knnb, scb, pw[4], sb[4], pw[5], sb[5], pw[6], sb[6], ftsA);

  // block 2: coords = fts (128 ch)
  knn_kernel<128><<<BATCH, NPTS, 0, stream>>>(ftsA, shift, knnb);
  shortcut_kernel<128, 256><<<NP / 64, 128, 0, stream>>>(ftsA, pw[11], sb[11], scb);
  edgeconv_kernel<128, 256><<<NP / 4, 128, 0, stream>>>(
      ftsA, knnb, scb, pw[8], sb[8], pw[9], sb[9], pw[10], sb[10], ftsB);

  pool_kernel<<<BATCH, 256, 0, stream>>>(ftsB, mask, pool);
  head_kernel<<<BATCH, 256, 0, stream>>>(pool, (const float*)d_in[65], (const float*)d_in[66],
                                         (const float*)d_in[67], (const float*)d_in[68],
                                         (float*)d_out);
}
